// BasicNCA_43508018709230
// MI455X (gfx1250) — matrix operand order repaired and checked by reading, not compiled, not run on hardware
//
#include <hip/hip_runtime.h>

#define NCA_C     24
#define NCA_HID   64
#define NCA_H     128
#define NCA_W     128
#define NCA_B     4
#define NCA_STEPS 32
#define PCH       72    // 3*C perception channels
#define K1PAD     96    // 72 padded to 3 K-tiles of 32
#define N2PAD     32    // 24 padded to 2 N-tiles of 16

typedef _Float16 half16 __attribute__((ext_vector_type(16)));
typedef _Float16 half8  __attribute__((ext_vector_type(8)));
typedef float    float8 __attribute__((ext_vector_type(8)));

// ---------------------------------------------------------------------------
// One-time (per launch) weight prep: f32 -> f16, transposed to [N][K] with
// zero padding, so every WMMA B fragment is one aligned 32B vector load.
//   w1h[n][k] : 64 x 96  (k >= 72 zero)
//   w2h[n][k] : 32 x 64  (n >= 24 zero)
// ---------------------------------------------------------------------------
__global__ void nca_prep_weights(const float* __restrict__ w1,
                                 const float* __restrict__ w2,
                                 _Float16* __restrict__ w1h,
                                 _Float16* __restrict__ w2h) {
  int i = blockIdx.x * blockDim.x + threadIdx.x;
  if (i < NCA_HID * K1PAD) {
    int n = i / K1PAD, k = i % K1PAD;
    w1h[i] = (k < PCH) ? (_Float16)w1[n * PCH + k] : (_Float16)0.0f;
  } else {
    int j = i - NCA_HID * K1PAD;
    if (j < N2PAD * NCA_HID) {
      int n = j / NCA_HID, k = j % NCA_HID;
      w2h[j] = (n < NCA_C) ? (_Float16)w2[n * NCA_HID + k] : (_Float16)0.0f;
    }
  }
}

// ---------------------------------------------------------------------------
// One NCA step. Workgroup = 128 threads (4 waves) owning a 16x16 pixel tile
// of one batch image. Each wave processes 4 image rows (M-tiles of 16 px).
// Per M-tile: GEMM1 16x96x64 (12 WMMA) -> relu -> GEMM2 16x64x32 (4 WMMA).
// ---------------------------------------------------------------------------
__launch_bounds__(128)
__global__ void nca_step(const float* __restrict__ src,
                         float* __restrict__ dst,
                         const _Float16* __restrict__ w1h,
                         const _Float16* __restrict__ w2h,
                         const float* __restrict__ b1,
                         const float* __restrict__ b2,
                         const float* __restrict__ mask) {
  __shared__ float    tile[NCA_C][18][20];           // halo'd state tile (34.5 KB)
  __shared__ _Float16 percep[4][16 * K1PAD];         // per-wave A staging (12 KB)
  __shared__ _Float16 hstage[4][16 * NCA_HID];       // per-wave hidden staging (8 KB)

  const int tid  = threadIdx.x;
  const int lane = tid & 31;
  const int wave = tid >> 5;
  const int g    = lane >> 4;   // wave32 half: selects K/M sub-block per ISA layout
  const int lm   = lane & 15;

  const int bid = blockIdx.x;
  const int tx  = bid & 7;
  const int ty  = (bid >> 3) & 7;
  const int b   = bid >> 6;
  const int y0  = ty * 16 - 1;
  const int x0  = tx * 16 - 1;

  // Touch the fire-mask rows early (lowers to global_prefetch_b8).
  if (tid < 16)
    __builtin_prefetch(&mask[((size_t)b * NCA_H + ty * 16 + tid) * NCA_W + tx * 16], 0, 3);

  // Cooperative load of the 18x18 halo tile for all 24 channels (zero pad = SAME).
  for (int i = tid; i < NCA_C * 18 * 18; i += 128) {
    int c  = i / (18 * 18);
    int r  = i % (18 * 18);
    int ly = r / 18, lx = r % 18;
    int gy = y0 + ly, gx = x0 + lx;
    float v = 0.0f;
    if (gy >= 0 && gy < NCA_H && gx >= 0 && gx < NCA_W)
      v = src[(((size_t)b * NCA_C + c) * NCA_H + gy) * NCA_W + gx];
    tile[c][ly][lx] = v;
  }
  __syncthreads();

  // Layer-2 B fragments live in registers for the whole tile (4 x 8 VGPRs).
  half16 B2f[2][2];
#pragma unroll
  for (int kt = 0; kt < 2; ++kt)
#pragma unroll
    for (int nt = 0; nt < 2; ++nt)
      B2f[kt][nt] = *(const half16*)(w2h + (nt * 16 + lm) * NCA_HID + kt * 32 + g * 16);

  float b1v[4];
#pragma unroll
  for (int nt = 0; nt < 4; ++nt) b1v[nt] = b1[nt * 16 + lm];
  float b2v[2];
#pragma unroll
  for (int nt = 0; nt < 2; ++nt) {
    int ch = nt * 16 + lm;
    b2v[nt] = (ch < NCA_C) ? b2[ch] : 0.0f;
  }

  _Float16* myP = percep[wave];
  _Float16* myH = hstage[wave];

  for (int it = 0; it < 4; ++it) {
    const int mtile = wave * 4 + it;   // image row within the 16x16 tile

    // ---- perception: identity / sobel-x / sobel-y -> f16 A staging --------
    for (int idx = lane; idx < 16 * K1PAD; idx += 32) {
      int m = idx / K1PAD;             // column within the row (= M index)
      int k = idx % K1PAD;
      _Float16 v = (_Float16)0.0f;
      if (k < PCH) {
        int c = k / 3, j = k - c * 3;
        const float* t = &tile[c][mtile + 1][m + 1];   // row stride 20
        float r;
        if (j == 0)      r = t[0];
        else if (j == 1) r = (-t[-21] + t[-19] - 2.0f * t[-1] + 2.0f * t[1]
                              - t[19] + t[21]) * 0.125f;
        else             r = (-t[-21] - 2.0f * t[-20] - t[-19]
                              + t[19] + 2.0f * t[20] + t[21]) * 0.125f;
        v = (_Float16)r;
      }
      myP[m * K1PAD + k] = v;
    }
    asm volatile("s_wait_dscnt 0x0" ::: "memory");  // wave-local LDS store->load

    // ---- GEMM1: (16 x 96) * (96 x 64) -> 16 x 64, f32 accumulate ----------
    float8 acc1[4];
#pragma unroll
    for (int nt = 0; nt < 4; ++nt)
#pragma unroll
      for (int r = 0; r < 8; ++r) acc1[nt][r] = 0.0f;

#pragma unroll
    for (int kt = 0; kt < 3; ++kt) {
      union { half16 v; half8 h[2]; } a;
      // A fragment, contiguous k order (same as B): lane half g holds K = kt*32 + g*16 + {0..15}
      a.h[0] = *(const half8*)(myP + lm * K1PAD + kt * 32 + g * 16);
      a.h[1] = *(const half8*)(myP + lm * K1PAD + kt * 32 + 8 + g * 16);
#pragma unroll
      for (int nt = 0; nt < 4; ++nt) {
        // B fragment, k order matches A: lane lm = N, 16 consecutive K at kt*32 + g*16
        half16 bf = *(const half16*)(w1h + (nt * 16 + lm) * K1PAD + kt * 32 + g * 16);
        acc1[nt] = __builtin_amdgcn_wmma_f32_16x16x32_f16(
            false, a.v, false, bf, (short)0, acc1[nt], false, false);
      }
    }

    // ---- bias + relu -> f16 hidden staging (A-layout source for GEMM2) ----
#pragma unroll
    for (int nt = 0; nt < 4; ++nt)
#pragma unroll
      for (int r = 0; r < 8; ++r) {
        float hv = acc1[nt][r] + b1v[nt];
        hv = hv > 0.0f ? hv : 0.0f;
        int m = r + g * 8;             // C/D layout: VGPR r, half g -> M = r + 8g
        myH[m * NCA_HID + nt * 16 + lm] = (_Float16)hv;
      }
    asm volatile("s_wait_dscnt 0x0" ::: "memory");

    // ---- GEMM2: (16 x 64) * (64 x 32) -> 16 x 32 (channels 0..23 valid) ---
    float8 acc2[2];
#pragma unroll
    for (int nt = 0; nt < 2; ++nt)
#pragma unroll
      for (int r = 0; r < 8; ++r) acc2[nt][r] = 0.0f;

#pragma unroll
    for (int kt = 0; kt < 2; ++kt) {
      union { half16 v; half8 h[2]; } a;
      a.h[0] = *(const half8*)(myH + lm * NCA_HID + kt * 32 + g * 16);
      a.h[1] = *(const half8*)(myH + lm * NCA_HID + kt * 32 + 8 + g * 16);
#pragma unroll
      for (int nt = 0; nt < 2; ++nt)
        acc2[nt] = __builtin_amdgcn_wmma_f32_16x16x32_f16(
            false, a.v, false, B2f[kt][nt], (short)0, acc2[nt], false, false);
    }

    // ---- stochastic-fire residual update ----------------------------------
#pragma unroll
    for (int nt = 0; nt < 2; ++nt) {
      int ch = nt * 16 + lm;
      if (ch < NCA_C) {
#pragma unroll
        for (int r = 0; r < 8; ++r) {
          int m  = r + g * 8;
          int gy = ty * 16 + mtile;
          int gx = tx * 16 + m;
          float fm   = mask[((size_t)b * NCA_H + gy) * NCA_W + gx];
          float fire = (fm < 0.5f) ? 1.0f : 0.0f;
          float sv   = tile[ch][mtile + 1][m + 1];
          dst[(((size_t)b * NCA_C + ch) * NCA_H + gy) * NCA_W + gx] =
              sv + (acc2[nt][r] + b2v[nt]) * fire;
        }
      }
    }
  }
}

// ---------------------------------------------------------------------------
// Host-side orchestration: weight prep + 32 ping-pong step launches.
// Workspace layout: [buf0 f32 state][buf1 f32 state][w1h f16][w2h f16]
// ---------------------------------------------------------------------------
extern "C" void kernel_launch(void* const* d_in, const int* in_sizes, int n_in,
                              void* d_out, int out_size, void* d_ws, size_t ws_size,
                              hipStream_t stream) {
  const float* state = (const float*)d_in[0];
  const float* w1    = (const float*)d_in[1];
  const float* b1    = (const float*)d_in[2];
  const float* w2    = (const float*)d_in[3];
  const float* b2    = (const float*)d_in[4];
  const float* masks = (const float*)d_in[5];
  float* out = (float*)d_out;

  const size_t stateElems = (size_t)NCA_B * NCA_C * NCA_H * NCA_W;
  float*    buf0 = (float*)d_ws;
  float*    buf1 = buf0 + stateElems;
  _Float16* w1h  = (_Float16*)(buf1 + stateElems);
  _Float16* w2h  = w1h + NCA_HID * K1PAD;

  {
    int total = NCA_HID * K1PAD + N2PAD * NCA_HID;
    nca_prep_weights<<<(total + 255) / 256, 256, 0, stream>>>(w1, w2, w1h, w2h);
  }

  dim3 grid(NCA_B * 8 * 8);   // 4 images x (128/16)^2 tiles
  dim3 block(128);            // 4 waves (wave32)
  for (int s = 0; s < NCA_STEPS; ++s) {
    const float* sp = (s == 0) ? state : ((s & 1) ? buf0 : buf1);
    float*       dp = (s == NCA_STEPS - 1) ? out : ((s & 1) ? buf1 : buf0);
    nca_step<<<grid, block, 0, stream>>>(
        sp, dp, w1h, w2h, b1, b2,
        masks + (size_t)s * NCA_B * NCA_H * NCA_W);
  }
}
